// AttentionWithContext_6055903887777
// MI455X (gfx1250) — compile-verified, matrix operand order checked by reading, not run on hardware
//
#include <hip/hip_runtime.h>
#include <hip/hip_bf16.h>

typedef _Float16 v16h __attribute__((ext_vector_type(16)));
typedef _Float16 v8h  __attribute__((ext_vector_type(8)));
typedef float    v8f  __attribute__((ext_vector_type(8)));
typedef float    v4f  __attribute__((ext_vector_type(4)));

#define FDIM 256
#define TDIM 2048
#define BDIM 128
#define LDS_STRIDE 264   // 256 + 8 halves pad -> 4-bank rotate per row, conflict-free B loads
#define TCHUNKS 8

#if __has_builtin(__builtin_amdgcn_tanhf)
  #define FAST_TANH(v) __builtin_amdgcn_tanhf(v)
#elif __has_builtin(__builtin_amdgcn_tanh_f32)
  #define FAST_TANH(v) __builtin_amdgcn_tanh_f32(v)
#else
  #define FAST_TANH(v) tanhf(v)
#endif

// ---------------------------------------------------------------------------
// Kernel 1: scores[row] = tanh(x[row,:] @ W + b) . u      (fused GEMM epilogue)
// 8 waves/WG; each wave owns TWO 16-row strips (32 rows): every B fragment
// feeds two WMMAs (halves LDS bandwidth per matrix op). B fragments are
// software-pipelined depth-2 across the flattened (n,k) space so the consumed
// fragment was loaded two steps earlier (s_wait_dscnt > 0, latency hidden).
// W^T staged once per WG in LDS as f16 (132 KB of the 320 KB WGP LDS).
// ---------------------------------------------------------------------------
__global__ __launch_bounds__(256) void awc_scores(const float* __restrict__ x,
                                                  const float* __restrict__ W,
                                                  const float* __restrict__ bias,
                                                  const float* __restrict__ u,
                                                  float* __restrict__ ait)
{
    __shared__ _Float16 sW[FDIM * LDS_STRIDE]; // sW[n*LDS_STRIDE + k] = (f16) W[k][n]

    const int tid = threadIdx.x;

    // Stage W^T into LDS: thread g=tid handles column g; pack pairs along f so
    // each iteration is a coalesced pair of global loads + one ds_store_b32.
    {
        _Float16* dst = &sW[(unsigned)tid * LDS_STRIDE];
        for (int f = 0; f < FDIM; f += 2) {
            float w0 = W[f * FDIM + tid];
            float w1 = W[(f + 1) * FDIM + tid];
            v8h p = {};
            p[0] = (_Float16)w0;
            p[1] = (_Float16)w1;
            *(__attribute__((ext_vector_type(2))) _Float16*)(dst + f) =
                __builtin_shufflevector(p, p, 0, 1);
        }
    }
    __syncthreads();

    const int wave  = tid >> 5;
    const int lane  = tid & 31;
    const int lrow  = lane & 15;          // row within strip (A) / output column (B,C)
    const int khalf = (lane >> 4) * 8;    // ISA 16-bit A layout: hi lanes take K+8 block

    const long rowBase = (long)blockIdx.x * 256 + wave * 32;

    // Build the A fragments (16x32 f16) for both strips; register resident.
    v16h afrag[2][8];
#pragma unroll
    for (int s = 0; s < 2; ++s) {
        const float* xr = x + (rowBase + s * 16 + lrow) * (long)FDIM;
#pragma unroll
        for (int k = 0; k < 8; ++k) {
            const int kb = k * 32 + khalf;
            v4f lo0 = *(const v4f*)(xr + kb);
            v4f lo1 = *(const v4f*)(xr + kb + 4);
            v4f hi0 = *(const v4f*)(xr + kb + 16);
            v4f hi1 = *(const v4f*)(xr + kb + 20);
            v16h a;
#pragma unroll
            for (int e = 0; e < 4; ++e) {
                a[e]      = (_Float16)lo0[e];
                a[e + 4]  = (_Float16)lo1[e];
                a[e + 8]  = (_Float16)hi0[e];
                a[e + 12] = (_Float16)hi1[e];
            }
            afrag[s][k] = a;
        }
    }

    float part[2][8];
#pragma unroll
    for (int s = 0; s < 2; ++s)
#pragma unroll
        for (int r = 0; r < 8; ++r) part[s][r] = 0.0f;

    // B fragment fetch for flattened step nk = n*8 + k (all indices constant
    // after unrolling).
    auto loadB = [&](int nk) -> v16h {
        const int nn = nk >> 3;
        const int kk = nk & 7;
        const _Float16* p = &sW[(nn * 16 + lrow) * LDS_STRIDE + kk * 32 + khalf];
        v8h blo = *(const v8h*)(p);          // ds_load_b128
        v8h bhi = *(const v8h*)(p + 16);     // ds_load_b128
        return __builtin_shufflevector(blo, bhi,
            0,1,2,3,4,5,6,7,8,9,10,11,12,13,14,15);
    };

    // Depth-2 rotating pipeline of B fragments.
    v16h bf0 = loadB(0);
    v16h bf1 = loadB(1);

#pragma unroll
    for (int n = 0; n < 16; ++n) {
        v8f acc0 = {};
        v8f acc1 = {};
#pragma unroll
        for (int k = 0; k < 8; ++k) {
            const int nk = n * 8 + k;
            v16h bcur = bf0;
            bf0 = bf1;
            if (nk + 2 < 128) bf1 = loadB(nk + 2);
            acc0 = __builtin_amdgcn_wmma_f32_16x16x32_f16(
                false, afrag[0][k], false, bcur, (short)0, acc0, false, false);
            acc1 = __builtin_amdgcn_wmma_f32_16x16x32_f16(
                false, afrag[1][k], false, bcur, (short)0, acc1, false, false);
        }
        const int col = n * 16 + lrow;                 // this lane's output column
        const float un = u[col];
        const float bn = bias[col];
        // C layout: lane = column, VGPR r = row r (lanes 0-15) / row r+8 (lanes 16-31)
#pragma unroll
        for (int r = 0; r < 8; ++r) {
            part[0][r] += FAST_TANH(acc0[r] + bn) * un;
            part[1][r] += FAST_TANH(acc1[r] + bn) * un;
        }
    }

    // Reduce over the 16 columns held by each lane half (xor-shuffle stays in-half)
#pragma unroll
    for (int s = 0; s < 2; ++s) {
#pragma unroll
        for (int r = 0; r < 8; ++r) {
            float v = part[s][r];
            v += __shfl_xor(v, 1, 32);
            v += __shfl_xor(v, 2, 32);
            v += __shfl_xor(v, 4, 32);
            v += __shfl_xor(v, 8, 32);
            part[s][r] = v;
        }
    }

    if (lrow == 0) {
#pragma unroll
        for (int s = 0; s < 2; ++s) {
            const long base = rowBase + s * 16 + (lane >> 4) * 8;
#pragma unroll
            for (int r = 0; r < 8; ++r) ait[base + r] = part[s][r];
        }
    }
}

// ---------------------------------------------------------------------------
// Kernel 2: a[b,t] = exp(ait[b,t]) / (sum_t exp(ait[b,t]) + 1e-7)  (in place)
// ---------------------------------------------------------------------------
__global__ __launch_bounds__(256) void awc_softmax(float* __restrict__ a)
{
    __shared__ float red[8];
    const int b   = blockIdx.x;
    const int tid = threadIdx.x;
    float* row = a + (long)b * TDIM;

    float s = 0.0f;
    for (int t = tid; t < TDIM; t += 256) s += expf(row[t]);
#pragma unroll
    for (int off = 16; off; off >>= 1) s += __shfl_xor(s, off, 32);
    if ((tid & 31) == 0) red[tid >> 5] = s;
    __syncthreads();

    float tot = 0.0f;
#pragma unroll
    for (int w = 0; w < 8; ++w) tot += red[w];
    const float inv = 1.0f / (tot + 1e-7f);

    for (int t = tid; t < TDIM; t += 256) row[t] = expf(row[t]) * inv;
}

// ---------------------------------------------------------------------------
// Kernel 3: partial weighted sums over T chunks (coalesced x re-read)
// ---------------------------------------------------------------------------
__global__ __launch_bounds__(256) void awc_wsum_partial(const float* __restrict__ x,
                                                        const float* __restrict__ a,
                                                        float* __restrict__ partial)
{
    const int b = blockIdx.y;
    const int c = blockIdx.x;            // 0..TCHUNKS-1
    const int f = threadIdx.x;
    const long tbase = (long)b * TDIM + (long)c * (TDIM / TCHUNKS);

    const float* xp = x + tbase * FDIM + f;
    const float* ap = a + tbase;
    float acc = 0.0f;
    for (int t = 0; t < TDIM / TCHUNKS; ++t)
        acc += ap[t] * xp[(long)t * FDIM];
    partial[((long)b * TCHUNKS + c) * FDIM + f] = acc;
}

// ---------------------------------------------------------------------------
// Kernel 4: deterministic reduction of partials -> out[b,f]
// ---------------------------------------------------------------------------
__global__ __launch_bounds__(256) void awc_wsum_reduce(const float* __restrict__ partial,
                                                       float* __restrict__ out)
{
    const int b = blockIdx.x;
    const int f = threadIdx.x;
    float acc = 0.0f;
#pragma unroll
    for (int c = 0; c < TCHUNKS; ++c)
        acc += partial[((long)b * TCHUNKS + c) * FDIM + f];
    out[(long)b * FDIM + f] = acc;
}

extern "C" void kernel_launch(void* const* d_in, const int* in_sizes, int n_in,
                              void* d_out, int out_size, void* d_ws, size_t ws_size,
                              hipStream_t stream) {
    const float* x    = (const float*)d_in[0];   // [B,T,F]
    const float* W    = (const float*)d_in[1];   // [F,F]
    const float* bias = (const float*)d_in[2];   // [F]
    const float* u    = (const float*)d_in[3];   // [F]
    float* out = (float*)d_out;                  // [B,F]

    float* a       = (float*)d_ws;                       // [B,T] scores -> weights
    float* partial = a + (size_t)BDIM * TDIM;            // [B,TCHUNKS,F]

    awc_scores      <<<dim3((BDIM * TDIM) / 256), 256, 0, stream>>>(x, W, bias, u, a);
    awc_softmax     <<<dim3(BDIM),                256, 0, stream>>>(a);
    awc_wsum_partial<<<dim3(TCHUNKS, BDIM),       256, 0, stream>>>(x, a, partial);
    awc_wsum_reduce <<<dim3(BDIM),                256, 0, stream>>>(partial, out);
}
